// RNN_27255862460778
// MI455X (gfx1250) — matrix operand order repaired and checked by reading, not compiled, not run on hardware
//
#include <hip/hip_runtime.h>
#include <hip/hip_bf16.h>
#include <math.h>

// ---------------------------------------------------------------------------
// MI455X (gfx1250) RNN LM pipeline.
//  - Xp / unembed GEMMs: exact f32 via V_WMMA_F32_16X16X4_F32; panels staged
//    with GLOBAL_LOAD_ASYNC_TO_LDS_B128 (+ s_wait_asynccnt).
//  - recurrence: persistent 1-WGP kernel, Wh resident in bf16:
//      256 KB in LDS (loaded by one TDM tensor_load_to_lds) + 128 VGPRs/wave,
//      V_WMMA_F32_16X16X32_BF16, ping-pong A-fragment h buffers,
//      one barrier per timestep.
// ---------------------------------------------------------------------------

typedef __attribute__((ext_vector_type(2)))  float  v2f;
typedef __attribute__((ext_vector_type(8)))  float  v8f;
typedef __attribute__((ext_vector_type(16))) __bf16 v16bf;

#define LAT   512
#define SEQT  1024
#define NBAT  4

#if __has_builtin(__builtin_amdgcn_tensor_load_to_lds)
#define HAVE_TDM 1
typedef __attribute__((ext_vector_type(4))) unsigned tdm_v4u;
typedef __attribute__((ext_vector_type(8))) int      tdm_v8i;
typedef __attribute__((ext_vector_type(4))) int      tdm_v4i;
#else
#define HAVE_TDM 0
#endif

// async global->LDS 16-byte copy (per active lane)
__device__ __forceinline__ void async_g2l_b128(void* lds, const void* gsrc) {
    unsigned           ldsa = (unsigned)(unsigned long long)lds;   // low 32 = LDS byte addr
    unsigned long long ga   = (unsigned long long)gsrc;
    asm volatile("global_load_async_to_lds_b128 %0, %1, off"
                 :: "v"(ldsa), "v"(ga) : "memory");
}
__device__ __forceinline__ void wait_async0() {
    asm volatile("s_wait_asynccnt 0x0" ::: "memory");
}

// ---------------------------------------------------------------------------
// prep: Wh (f32, row-major 512x512, K x N) -> bf16 WMMA B-fragment layout.
// Unit layout: frag[((ntile*16 + ktile)*32 + lane)*16 + i]
//   element = Wh[ ktile*32 + (i&8)*2 + (lane>>4)*8 + (i&7) ][ ntile*16 + (lane&15) ]
// ---------------------------------------------------------------------------
__global__ void prep_whfrag(const float* __restrict__ Wh, __bf16* __restrict__ Wf) {
    int o = blockIdx.x * blockDim.x + threadIdx.x;
    if (o >= LAT * LAT) return;
    int i    =  o        & 15;
    int lane = (o >> 4)  & 31;
    int kt   = (o >> 9)  & 15;
    int nt   =  o >> 13;
    int krow = kt * 32 + ((i & 8) * 2) + ((lane >> 4) * 8) + (i & 7);
    int col  = nt * 16 + (lane & 15);
    Wf[o] = (__bf16)Wh[krow * LAT + col];
}

// ---------------------------------------------------------------------------
// Tiled f32 GEMM: C[M x N] = A[M x 512] * B[512 x N] + bias, M = grid.x*128.
// GATHER: A row r = embed[tok[r]].  Block: 256 thr = 8 waves (2M x 4N),
// wave tile 64x32 via 4x2 accumulators, K panels of 32 staged async into LDS.
// ---------------------------------------------------------------------------
template<bool GATHER>
__global__ __launch_bounds__(256)
void gemm_f32_wmma(const float* __restrict__ A, const int* __restrict__ tok,
                   const float* __restrict__ Bw, const float* __restrict__ bias,
                   float* __restrict__ C, int N) {
    __shared__ float As[128][36];   // 144B rows: 16B-aligned float4 slots
    __shared__ float Bs[32][128];

    const int tid  = threadIdx.x;
    const int lane = tid & 31;
    const int w    = tid >> 5;
    const int wm   = w >> 2;        // 0..1
    const int wn   = w & 3;         // 0..3
    const int m0   = blockIdx.x * 128;
    const int n0   = blockIdx.y * 128;

    v8f acc[4][2] = {};

    for (int kb = 0; kb < LAT; kb += 32) {
        // ---- stage A panel (128 x 32) via async global->LDS ----
#pragma unroll
        for (int it = 0; it < 4; ++it) {
            int idx = tid + it * 256;           // 0..1023 float4 slots
            int row = idx >> 3;
            int c4  = (idx & 7) * 4;
            int gr  = m0 + row;
            const float* src = GATHER
                ? (A + (size_t)tok[gr] * LAT + kb + c4)
                : (A + (size_t)gr     * LAT + kb + c4);
            async_g2l_b128(&As[row][c4], src);
        }
        // ---- stage B panel (32 x 128), clamp N tail ----
#pragma unroll
        for (int it = 0; it < 4; ++it) {
            int idx = tid + it * 256;
            int row = idx >> 5;
            int c4  = (idx & 31) * 4;
            int gk  = kb + row;
            int gn  = n0 + c4;
            if (gn + 3 < N) {
                async_g2l_b128(&Bs[row][c4], Bw + (size_t)gk * N + gn);
            } else {
#pragma unroll
                for (int j = 0; j < 4; ++j) {
                    int cc = gn + j; if (cc > N - 1) cc = N - 1;
                    Bs[row][c4 + j] = Bw[(size_t)gk * N + cc];
                }
            }
        }
        wait_async0();
        __syncthreads();

        // ---- 8 k-steps of K=4 f32 WMMA ----
#pragma unroll
        for (int ks = 0; ks < 8; ++ks) {
            int k0 = ks * 4 + ((lane >> 4) << 1);   // lane-half picks K pair
            v2f a[4], b[2];
#pragma unroll
            for (int mt = 0; mt < 4; ++mt) {
                const float* p = &As[wm * 64 + mt * 16 + (lane & 15)][k0];
                a[mt][0] = p[0]; a[mt][1] = p[1];
            }
#pragma unroll
            for (int nt = 0; nt < 2; ++nt) {
                int nn = wn * 32 + nt * 16 + (lane & 15);
                b[nt][0] = Bs[k0 + 0][nn];
                b[nt][1] = Bs[k0 + 1][nn];
            }
#pragma unroll
            for (int mt = 0; mt < 4; ++mt)
#pragma unroll
                for (int nt = 0; nt < 2; ++nt)
                    acc[mt][nt] = __builtin_amdgcn_wmma_f32_16x16x4_f32(
                        false, a[mt], false, b[nt], (short)0, acc[mt][nt],
                        false, false);
        }
        __syncthreads();
    }

    // ---- store C (+bias, N-masked). D layout: vgpr r -> M=r (lanes<16) / 8+r ----
#pragma unroll
    for (int mt = 0; mt < 4; ++mt)
#pragma unroll
        for (int nt = 0; nt < 2; ++nt) {
            int col = n0 + wn * 32 + nt * 16 + (lane & 15);
            if (col < N) {
                float bv = bias ? bias[col] : 0.0f;
                int rbase = m0 + wm * 64 + mt * 16 + ((lane >= 16) ? 8 : 0);
#pragma unroll
                for (int r = 0; r < 8; ++r)
                    C[(size_t)(rbase + r) * N + col] = acc[mt][nt][r] + bv;
            }
        }
}

// ---------------------------------------------------------------------------
// Persistent RNN scan: one workgroup, 512 thr = 16 waves, each wave owns
// 32 output columns (n-tiles 2w, 2w+1). Wh bf16 fragments: ktile 0..7 in LDS
// (256 KB, DMA'd by the Tensor Data Mover), ktile 8..15 in VGPRs (128/wave).
// h kept as bf16 A-fragments in LDS, ping-pong; one barrier per timestep.
// ---------------------------------------------------------------------------
__global__ __launch_bounds__(512)
void rnn_scan(const float* __restrict__ Xp, const __bf16* __restrict__ Wfrag,
              float* __restrict__ Hout) {
    extern __shared__ char smem[];
    v16bf*  ldsW  = (v16bf*)smem;                        // 32nt*8kt*32lane units = 256KB
    __bf16* ldsH  = (__bf16*)(smem + 32 * 8 * 32 * 32);  // 2buf*16kt*32lane*16 = 32KB
    v16bf*  ldsHv = (v16bf*)ldsH;

    const int lane = threadIdx.x & 31;
    const int w    = threadIdx.x >> 5;      // 0..15

    // zero both h-fragment buffers
    {
        uint32_t* p = (uint32_t*)ldsH;
        for (int i = threadIdx.x; i < 8192; i += 512) p[i] = 0u;
    }

    const v16bf* Wg = (const v16bf*)Wfrag;  // unit idx ((nt*16+kt)*32+lane)

#if HAVE_TDM
    // TDM: 2D tile, data_size=8B. 32 rows (n-tiles) of 1024 units (8KB =
    // ktiles 0..7), global row stride 2048 units (16KB) -> packed into LDS.
    if (w == 0) {
        unsigned           la = (unsigned)(unsigned long long)(void*)ldsW;
        unsigned long long ga = (unsigned long long)(const void*)Wfrag;
        tdm_v4u g0 = { 1u,                                   // count=1 (valid user D#)
                       la,                                   // lds_addr
                       (unsigned)ga,                         // global_addr[31:0]
                       ((unsigned)(ga >> 32) & 0x1FFFFFFu)   // global_addr[56:32]
                         | (2u << 30) };                     // type=2 ("image")
        tdm_v8i g1 = { (int)(3u << 16),        // wg_mask=0 | data_size=3 (8B)
                       (int)(1024u << 16),     // tensor_dim0=1024 (lo16 in [31:16])
                       (int)(32u  << 16),      // tensor_dim0 hi=0 | tensor_dim1=32
                       (int)(1024u << 16),     // tensor_dim1 hi=0 | tile_dim0=1024
                       (int)32,                // tile_dim1=32 | tile_dim2=0
                       (int)2048,              // tensor_dim0_stride lo32 = 2048
                       0, 0 };                 // stride0 hi / tensor_dim1_stride = 0
        tdm_v4i gz = { 0, 0, 0, 0 };
#if __clang_major__ >= 23
        tdm_v8i gz8 = { 0, 0, 0, 0, 0, 0, 0, 0 };
        __builtin_amdgcn_tensor_load_to_lds(g0, g1, gz, gz, gz8, 0);
#else
        __builtin_amdgcn_tensor_load_to_lds(g0, g1, gz, gz, 0);
#endif
        __builtin_amdgcn_s_wait_tensorcnt(0);
    }
#else
    // fallback: manual copy of this wave's two n-tiles (ktiles 0..7)
#pragma unroll
    for (int ti = 0; ti < 2; ++ti) {
        int nt = 2 * w + ti;
#pragma unroll
        for (int kt = 0; kt < 8; ++kt)
            ldsW[(nt * 8 + kt) * 32 + lane] = Wg[(nt * 16 + kt) * 32 + lane];
    }
#endif

    // VGPR-resident half (ktiles 8..15 of this wave's two n-tiles)
    v16bf wreg[2][8];
#pragma unroll
    for (int ti = 0; ti < 2; ++ti) {
        int nt = 2 * w + ti;
#pragma unroll
        for (int kt = 0; kt < 8; ++kt)
            wreg[ti][kt] = Wg[(nt * 16 + 8 + kt) * 32 + lane];
    }
    __syncthreads();

    for (int t = 0; t < SEQT; ++t) {
        const int rbuf = t & 1, wbuf = rbuf ^ 1;
        v8f acc0 = {}, acc1 = {};

#pragma unroll
        for (int kt = 0; kt < 8; ++kt) {               // LDS-resident K 0..255
            v16bf af = ldsHv[(rbuf * 16 + kt) * 32 + lane];
            acc0 = __builtin_amdgcn_wmma_f32_16x16x32_bf16(
                false, af, false, ldsW[((2 * w + 0) * 8 + kt) * 32 + lane],
                (short)0, acc0, false, false);
            acc1 = __builtin_amdgcn_wmma_f32_16x16x32_bf16(
                false, af, false, ldsW[((2 * w + 1) * 8 + kt) * 32 + lane],
                (short)0, acc1, false, false);
        }
#pragma unroll
        for (int kt = 0; kt < 8; ++kt) {               // VGPR-resident K 256..511
            v16bf af = ldsHv[(rbuf * 16 + 8 + kt) * 32 + lane];
            acc0 = __builtin_amdgcn_wmma_f32_16x16x32_bf16(
                false, af, false, wreg[0][kt], (short)0, acc0, false, false);
            acc1 = __builtin_amdgcn_wmma_f32_16x16x32_bf16(
                false, af, false, wreg[1][kt], (short)0, acc1, false, false);
        }

        // epilogue: rows M=0..3 (= batch) live in vgprs 0..3 of lanes 0..15
        if (lane < 16) {
#pragma unroll
            for (int ti = 0; ti < 2; ++ti) {
                int n     = w * 32 + ti * 16 + lane;
                int kt    = n >> 5;
                int kk    = n & 31;
                int khalf = (kk >> 3) & 1;
                int j     = (kk & 16) ? (4 + ((kk - 16 - khalf * 8) >> 1))
                                      : ((kk - khalf * 8) >> 1);
                int ii    = 2 * j + (kk & 1);
#pragma unroll
                for (int r = 0; r < 4; ++r) {          // r == batch index
                    float v  = ti ? acc1[r] : acc0[r];
                    size_t g = (size_t)(r * SEQT + t) * LAT + n;
                    float y  = tanhf(v + Xp[g]);
                    Hout[g]  = y;
                    // scatter into next step's A-fragment (M=r, K=n)
                    ldsH[((wbuf * 16 + kt) * 32 + (16 * khalf + r)) * 16 + ii] =
                        (__bf16)y;
                }
            }
        }
        __syncthreads();
    }
}

// ---------------------------------------------------------------------------
extern "C" void kernel_launch(void* const* d_in, const int* in_sizes, int n_in,
                              void* d_out, int out_size, void* d_ws, size_t ws_size,
                              hipStream_t stream) {
    const int*   x     = (const int*)  d_in[0];
    const float* embed = (const float*)d_in[1];
    const float* Wx0   = (const float*)d_in[2];
    const float* Wh0   = (const float*)d_in[3];
    const float* b0    = (const float*)d_in[4];
    const float* Wx1   = (const float*)d_in[5];
    const float* Wh1   = (const float*)d_in[6];
    const float* b1    = (const float*)d_in[7];
    const float* Wu    = (const float*)d_in[8];
    const float* bu    = (const float*)d_in[9];
    float* out = (float*)d_out;

    char* ws = (char*)d_ws;
    __bf16* Wf0 = (__bf16*)ws;                              // 512 KB
    __bf16* Wf1 = (__bf16*)(ws + (512 << 10));              // 512 KB
    float*  Xp  = (float*) (ws + (1 << 20));                // 8 MB
    float*  H   = (float*) (ws + (1 << 20) + (8 << 20));    // 8 MB

    const int SCAN_LDS = 288 * 1024;
    hipFuncSetAttribute((const void*)rnn_scan,
                        hipFuncAttributeMaxDynamicSharedMemorySize, SCAN_LDS);

    prep_whfrag<<<1024, 256, 0, stream>>>(Wh0, Wf0);
    prep_whfrag<<<1024, 256, 0, stream>>>(Wh1, Wf1);

    dim3 blk(256);
    dim3 g0(4096 / 128, 512 / 128);
    gemm_f32_wmma<true ><<<g0, blk, 0, stream>>>(embed, x, Wx0, b0, Xp, 512);
    rnn_scan<<<1, 512, SCAN_LDS, stream>>>(Xp, Wf0, H);
    gemm_f32_wmma<false><<<g0, blk, 0, stream>>>(H, nullptr, Wx1, b1, Xp, 512);
    rnn_scan<<<1, 512, SCAN_LDS, stream>>>(Xp, Wf1, H);

    dim3 g1(4096 / 128, (50257 + 127) / 128);
    gemm_f32_wmma<false><<<g1, blk, 0, stream>>>(H, nullptr, Wu, bu, out, 50257);
}
